// EncoderBlock_35588099015390
// MI455X (gfx1250) — matrix operand order repaired and checked by reading, not compiled, not run on hardware
//
#include <hip/hip_runtime.h>
#include <stdint.h>

// ---- problem constants (match reference) ----
#define Bb   2
#define Ss   2048
#define Ee   1024
#define Hh   16
#define FFf  4096
#define DHd  64
#define Mrows (Bb*Ss)   // 4096

typedef unsigned short u16;
typedef __attribute__((ext_vector_type(4)))  unsigned int   u32x4;
typedef __attribute__((ext_vector_type(8)))  unsigned short u16x8;
typedef __attribute__((ext_vector_type(4)))  unsigned short u16x4;
typedef __attribute__((ext_vector_type(4)))  float          f32x4;
typedef __attribute__((ext_vector_type(16))) __bf16         v16bf;
typedef __attribute__((ext_vector_type(8)))  float          v8f;

struct F2 { u32x4 a, b; };

// Load a 16-element bf16 fragment as two 16B chunks (chunk1 at p, chunk2 at p+step)
__device__ __forceinline__ v16bf frag_ld(const u16* p, int step) {
  F2 t;
  t.a = *(const u32x4*)(p);
  t.b = *(const u32x4*)(p + step);
  return __builtin_bit_cast(v16bf, t);
}

__device__ __forceinline__ u16 f2bf(float f) {
  unsigned u = __builtin_bit_cast(unsigned, f);
  u += 0x7fffu + ((u >> 16) & 1u);   // round-to-nearest-even
  return (u16)(u >> 16);
}

__device__ __forceinline__ v8f wmma_bf16(v16bf a, v16bf b, v8f c) {
  // 8 args: (neg_a, A, neg_b, B, c_mod, C, reuse_a, reuse_b)
  return __builtin_amdgcn_wmma_f32_16x16x32_bf16(false, a, false, b, (short)0, c, false, false);
}

// Async global->LDS copy of 16 bytes (CDNA5 GLOBAL_LOAD_ASYNC_TO_LDS_B128, GV mode).
// ldsaddr: LDS byte address (low 32 bits of a generic pointer to __shared__).
__device__ __forceinline__ void async_g2l_b128(unsigned ldsaddr, const void* gptr) {
  asm volatile("global_load_async_to_lds_b128 %0, %1, off"
               :: "v"(ldsaddr), "v"(gptr) : "memory");
}
__device__ __forceinline__ void wait_asynccnt0() {
  asm volatile("s_wait_asynccnt 0x0" ::: "memory");
}

// ---------------------------------------------------------------------------
// f32 -> bf16 elementwise conversion (weights)
// ---------------------------------------------------------------------------
__global__ void cvt_f32_bf16(const float* __restrict__ in, u16* __restrict__ out, long n) {
  long i = (long)blockIdx.x * blockDim.x + threadIdx.x;
  if (i < n) out[i] = f2bf(in[i]);
}

// ---------------------------------------------------------------------------
// LayerNorm (ddof=1 std, divide by (std+eps)); one block per row, E=1024
// ---------------------------------------------------------------------------
__global__ __launch_bounds__(256) void layernorm_bf16(
    const float* __restrict__ x, const float* __restrict__ gamma,
    const float* __restrict__ beta, u16* __restrict__ out)
{
  __shared__ float red[16];
  const int tid  = threadIdx.x;
  const int lane = tid & 31, wid = tid >> 5;
  const long row = blockIdx.x;
  const float* xr = x + row * (long)Ee;

  f32x4 v = *(const f32x4*)(xr + tid * 4);
  float s = v[0] + v[1] + v[2] + v[3];
  float q = v[0]*v[0] + v[1]*v[1] + v[2]*v[2] + v[3]*v[3];
#pragma unroll
  for (int m = 1; m < 32; m <<= 1) {
    s += __shfl_xor(s, m, 32);
    q += __shfl_xor(q, m, 32);
  }
  if (lane == 0) { red[wid*2] = s; red[wid*2+1] = q; }
  __syncthreads();
  float ts = 0.f, tq = 0.f;
#pragma unroll
  for (int w = 0; w < 8; ++w) { ts += red[w*2]; tq += red[w*2+1]; }
  const float mean = ts * (1.0f / (float)Ee);
  const float var  = (tq - ts * ts * (1.0f / (float)Ee)) * (1.0f / (float)(Ee - 1));
  const float inv  = 1.0f / (sqrtf(fmaxf(var, 0.f)) + 1e-6f);

  const int c = tid * 4;
  f32x4 g  = *(const f32x4*)(gamma + c);
  f32x4 be = *(const f32x4*)(beta + c);
  u16x4 o;
#pragma unroll
  for (int j = 0; j < 4; ++j)
    o[j] = f2bf(be[j] + g[j] * (v[j] - mean) * inv);
  *(u16x4*)(out + row * (long)Ee + c) = o;
}

// ---------------------------------------------------------------------------
// bf16 WMMA GEMM: out = epilogue(A[MxK] @ W[KxN])
//   block tile 128x128, 8 waves (4x2), each wave 32x64 (2x4 WMMA tiles), K step 32
//   A tile staged via async global->LDS (ASYNCcnt); W tile staged transposed.
//   SPLIT: write to [B,H,S,DH] (for Q/K/V); OBF: bf16 out, else f32
// ---------------------------------------------------------------------------
#define GMB 128
#define GNB 128
#define GKB 32
#define LDA 40
#define LDB 40

template<int SPLIT, int HASB, int RELU, int RESID, int OBF>
__global__ __launch_bounds__(256) void gemm_wmma(
    const u16* __restrict__ A, const u16* __restrict__ W,
    const float* __restrict__ bias, const float* __restrict__ resid,
    void* __restrict__ outp, int M, int N, int K)
{
  __shared__ u16 As[GMB * LDA];   // 128x32 row-major (padded)
  __shared__ u16 Bs[GNB * LDB];   // [n][k] transposed (padded)
  const int tid  = threadIdx.x;
  const int lane = tid & 31, wid = tid >> 5;
  const int half = lane >> 4, lr = lane & 15;
  const int wm = wid >> 1, wn = wid & 1;
  const long row0 = (long)blockIdx.x * GMB;
  const long col0 = (long)blockIdx.y * GNB;

  v8f z = {0.f,0.f,0.f,0.f,0.f,0.f,0.f,0.f};
  v8f acc[2][4] = {{z,z,z,z},{z,z,z,z}};

  for (int k0 = 0; k0 < K; k0 += GKB) {
    __syncthreads();
    // stage A tile 128x32 via async copies: 2 x 16B per thread, straight to LDS
    {
      int c = tid;
#pragma unroll
      for (int it = 0; it < 2; ++it, c += 256) {
        int r = c >> 2, kc = (c & 3) << 3;
        const u16* gp = A + (row0 + r) * (long)K + k0 + kc;
        unsigned lds = (unsigned)(uintptr_t)(As + r * LDA + kc);
        async_g2l_b128(lds, gp);
      }
    }
    // stage W tile 32x128 transposed -> Bs[n][k]  (2 x 16B per thread + scatter)
    {
      int c = tid;
#pragma unroll
      for (int it = 0; it < 2; ++it, c += 256) {
        int kr = c >> 4, nc = (c & 15) << 3;
        u32x4 d = *(const u32x4*)(W + (long)(k0 + kr) * N + col0 + nc);
        u16x8 s = __builtin_bit_cast(u16x8, d);
#pragma unroll
        for (int j = 0; j < 8; ++j) Bs[(nc + j) * LDB + kr] = s[j];
      }
    }
    if (k0 + GKB < K)
      __builtin_prefetch(W + (long)(k0 + GKB) * N + col0, 0, 3);
    wait_asynccnt0();          // drain this wave's async global->LDS copies
    __syncthreads();           // make all waves' tiles visible

    // A fragment: lane holds row lr, K chunks {half*8..+7, half*8+16..+23}
    v16bf a0 = frag_ld(As + (wm*32 +  0 + lr) * LDA + half*8, 16);
    v16bf a1 = frag_ld(As + (wm*32 + 16 + lr) * LDA + half*8, 16);
#pragma unroll
    for (int j = 0; j < 4; ++j) {
      // B fragment: lane holds col lr, K chunks {half*8..+7, half*8+16..+23}
      v16bf bj = frag_ld(Bs + (wn*64 + j*16 + lr) * LDB + half*8, 16);
      acc[0][j] = wmma_bf16(a0, bj, acc[0][j]);
      acc[1][j] = wmma_bf16(a1, bj, acc[1][j]);
    }
  }

#pragma unroll
  for (int i = 0; i < 2; ++i) {
#pragma unroll
    for (int j = 0; j < 4; ++j) {
      long col = col0 + wn*64 + j*16 + lr;
      float bv = HASB ? bias[col] : 0.f;
      long mb = row0 + wm*32 + i*16 + half*8;
#pragma unroll
      for (int r = 0; r < 8; ++r) {
        long m = mb + r;
        float vv = acc[i][j][r] + bv;
        if (RELU)  vv = fmaxf(vv, 0.f);
        if (RESID) vv += resid[m * (long)N + col];
        long idx;
        if (SPLIT) {
          long bbi = m >> 11, ssi = m & (Ss - 1);
          long hhi = col >> 6, ddi = col & (DHd - 1);
          idx = ((bbi * Hh + hhi) * (long)Ss + ssi) * DHd + ddi;
        } else {
          idx = m * (long)N + col;
        }
        if (OBF) ((u16*)outp)[idx]   = f2bf(vv);
        else     ((float*)outp)[idx] = vv;
      }
    }
  }
}

// ---------------------------------------------------------------------------
// Flash attention: grid (S/64, B*H), 128 threads = 4 waves, wave owns 16 q rows.
// Q,K,V bf16 [B,H,S,DH]; ctx bf16 written as [B,S,E] (head-merged).
// ---------------------------------------------------------------------------
__global__ __launch_bounds__(128) void attn_fwd(
    const u16* __restrict__ Q, const u16* __restrict__ Km, const u16* __restrict__ V,
    const int* __restrict__ mask, u16* __restrict__ ctx)
{
  __shared__ u16 Vt[64 * 40];       // V^T tile: [dh][key 0..31] padded
  __shared__ u16 Pl[4 * 16 * 40];   // per-wave P tile: [m 0..15][key 0..31] padded

  const int tid  = threadIdx.x;
  const int lane = tid & 31, wid = tid >> 5;
  const int half = lane >> 4, lr = lane & 15;
  const int bh = blockIdx.y;
  const int b = bh >> 4, h = bh & 15;
  const long base = (long)bh * Ss * DHd;
  const int q0 = blockIdx.x * 64 + wid * 16;

  // Q A-fragments (q rows x dh), dh split 0..31 / 32..63
  const u16* qrow = Q + base + (long)(q0 + lr) * DHd;
  const v16bf aq0 = frag_ld(qrow +      half*16, 8);
  const v16bf aq1 = frag_ld(qrow + 32 + half*16, 8);

  float mi[8], li[8];
#pragma unroll
  for (int r = 0; r < 8; ++r) { mi[r] = -3.0e38f; li[r] = 0.f; }
  v8f z = {0.f,0.f,0.f,0.f,0.f,0.f,0.f,0.f};
  v8f o[4] = {z, z, z, z};

  const int* mrow = mask + b * Ss;

  for (int kb = 0; kb < Ss; kb += 32) {
    __syncthreads();
    // stage V^T tile (32 keys x 64 dh -> Vt[dh][key])
#pragma unroll
    for (int it = 0; it < 2; ++it) {
      int c = tid + it * 128;
      int kr = c >> 3, dc = (c & 7) << 3;
      u32x4 d = *(const u32x4*)(V + base + (long)(kb + kr) * DHd + dc);
      u16x8 s = __builtin_bit_cast(u16x8, d);
#pragma unroll
      for (int j = 0; j < 8; ++j) Vt[(dc + j) * 40 + kr] = s[j];
    }
    __syncthreads();

    // scores: Q @ K^T for two 16-key subtiles (B columns = K rows, contiguous)
    v8f s0 = z, s1 = z;
    {
      const u16* k0p = Km + base + (long)(kb +      lr) * DHd;
      const u16* k1p = Km + base + (long)(kb + 16 + lr) * DHd;
      s0 = wmma_bf16(aq0, frag_ld(k0p +      half*16, 8), s0);
      s0 = wmma_bf16(aq1, frag_ld(k0p + 32 + half*16, 8), s0);
      s1 = wmma_bf16(aq0, frag_ld(k1p +      half*16, 8), s1);
      s1 = wmma_bf16(aq1, frag_ld(k1p + 32 + half*16, 8), s1);
    }
    const int mk0 = mrow[kb + lr];
    const int mk1 = mrow[kb + 16 + lr];

#pragma unroll
    for (int r = 0; r < 8; ++r) {
      float v0 = mk0 ? s0[r] * 0.125f : -1.0e9f;   // 1/sqrt(64)
      float v1 = mk1 ? s1[r] * 0.125f : -1.0e9f;
      float mx = fmaxf(v0, v1);
#pragma unroll
      for (int m = 1; m < 16; m <<= 1) mx = fmaxf(mx, __shfl_xor(mx, m, 32));
      float mnew = fmaxf(mi[r], mx);
      float al = __expf(mi[r] - mnew);
      float p0 = __expf(v0 - mnew);
      float p1 = __expf(v1 - mnew);
      float sm = p0 + p1;
#pragma unroll
      for (int m = 1; m < 16; m <<= 1) sm += __shfl_xor(sm, m, 32);
      li[r] = li[r] * al + sm;
      mi[r] = mnew;
      o[0][r] = o[0][r] * al;
      o[1][r] = o[1][r] * al;
      o[2][r] = o[2][r] * al;
      o[3][r] = o[3][r] * al;
      // C layout: this lane holds row m = r + 8*half, col = lr (+16 for s1)
      const int mrw = r + half * 8;
      Pl[(wid*16 + mrw) * 40 +      lr] = f2bf(p0);
      Pl[(wid*16 + mrw) * 40 + 16 + lr] = f2bf(p1);
    }
    // wave-internal LDS write->read: drain DScnt before re-reading P as A-fragment
    asm volatile("s_wait_dscnt 0x0" ::: "memory");

    v16bf ap = frag_ld(&Pl[(wid*16 + lr) * 40 + half*16], 8);
#pragma unroll
    for (int nt = 0; nt < 4; ++nt) {
      v16bf bv = frag_ld(&Vt[(nt*16 + lr) * 40 + half*16], 8);
      o[nt] = wmma_bf16(ap, bv, o[nt]);
    }
  }

  // finalize: divide by row sums, write ctx[b, s, h*64+dh] as bf16
#pragma unroll
  for (int r = 0; r < 8; ++r) {
    float inv = li[r] > 0.f ? 1.f / li[r] : 0.f;
    int mrw = r + half * 8;
    long srow = q0 + mrw;
    long idxb = ((long)b * Ss + srow) * Ee + (long)h * DHd + lr;
#pragma unroll
    for (int nt = 0; nt < 4; ++nt)
      ctx[idxb + nt * 16] = f2bf(o[nt][r] * inv);
  }
}

// ---------------------------------------------------------------------------
// Host launcher
// ---------------------------------------------------------------------------
extern "C" void kernel_launch(void* const* d_in, const int* in_sizes, int n_in,
                              void* d_out, int out_size, void* d_ws, size_t ws_size,
                              hipStream_t stream) {
  const float* x    = (const float*)d_in[0];
  const int*   mask = (const int*)d_in[1];
  const float* wq = (const float*)d_in[2];
  const float* wk = (const float*)d_in[3];
  const float* wv = (const float*)d_in[4];
  const float* wo = (const float*)d_in[5];
  const float* w1 = (const float*)d_in[6];
  const float* b1 = (const float*)d_in[7];
  const float* w2 = (const float*)d_in[8];
  const float* b2 = (const float*)d_in[9];
  const float* g1  = (const float*)d_in[10];
  const float* be1 = (const float*)d_in[11];
  const float* g2  = (const float*)d_in[12];
  const float* be2 = (const float*)d_in[13];
  float* out = (float*)d_out;

  char* ws = (char*)d_ws;
  size_t off = 0;
  auto alloc = [&](size_t bytes) -> char* {
    char* p = ws + off;
    off += (bytes + 255) & ~(size_t)255;
    return p;
  };

  u16* wq_b  = (u16*)alloc((size_t)Ee * Ee * 2);
  u16* wk_b  = (u16*)alloc((size_t)Ee * Ee * 2);
  u16* wv_b  = (u16*)alloc((size_t)Ee * Ee * 2);
  u16* wo_b  = (u16*)alloc((size_t)Ee * Ee * 2);
  u16* w1_b  = (u16*)alloc((size_t)Ee * FFf * 2);
  u16* w2_b  = (u16*)alloc((size_t)FFf * Ee * 2);
  u16* n1_b  = (u16*)alloc((size_t)Mrows * Ee * 2);
  u16* q_b   = (u16*)alloc((size_t)Mrows * Ee * 2);
  u16* k_b   = (u16*)alloc((size_t)Mrows * Ee * 2);
  u16* v_b   = (u16*)alloc((size_t)Mrows * Ee * 2);
  u16* ctx_b = (u16*)alloc((size_t)Mrows * Ee * 2);
  u16* n2_b  = (u16*)alloc((size_t)Mrows * Ee * 2);
  u16* ff1_b = (u16*)alloc((size_t)Mrows * FFf * 2);
  float* x1  = (float*)alloc((size_t)Mrows * Ee * 4);

  auto cvt = [&](const float* src, u16* dst, long n) {
    cvt_f32_bf16<<<dim3((unsigned)((n + 255) / 256)), dim3(256), 0, stream>>>(src, dst, n);
  };
  cvt(wq, wq_b, (long)Ee * Ee);
  cvt(wk, wk_b, (long)Ee * Ee);
  cvt(wv, wv_b, (long)Ee * Ee);
  cvt(wo, wo_b, (long)Ee * Ee);
  cvt(w1, w1_b, (long)Ee * FFf);
  cvt(w2, w2_b, (long)FFf * Ee);

  // n1 = layernorm(x)
  layernorm_bf16<<<dim3(Mrows), dim3(256), 0, stream>>>(x, g1, be1, n1_b);

  // Q/K/V projections -> [B,H,S,DH] bf16
  dim3 gE(Mrows / GMB, Ee / GNB);     // (32, 8)
  gemm_wmma<1,0,0,0,1><<<gE, 256, 0, stream>>>(n1_b, wq_b, nullptr, nullptr, q_b, Mrows, Ee, Ee);
  gemm_wmma<1,0,0,0,1><<<gE, 256, 0, stream>>>(n1_b, wk_b, nullptr, nullptr, k_b, Mrows, Ee, Ee);
  gemm_wmma<1,0,0,0,1><<<gE, 256, 0, stream>>>(n1_b, wv_b, nullptr, nullptr, v_b, Mrows, Ee, Ee);

  // attention -> ctx (head-merged [B,S,E] bf16)
  attn_fwd<<<dim3(Ss / 64, Bb * Hh), dim3(128), 0, stream>>>(q_b, k_b, v_b, mask, ctx_b);

  // x1 = x + ctx @ wo   (f32)
  gemm_wmma<0,0,0,1,0><<<gE, 256, 0, stream>>>(ctx_b, wo_b, nullptr, x, x1, Mrows, Ee, Ee);

  // n2 = layernorm(x1)
  layernorm_bf16<<<dim3(Mrows), dim3(256), 0, stream>>>(x1, g2, be2, n2_b);

  // ff1 = relu(n2 @ w1 + b1)  (bf16)
  dim3 gF(Mrows / GMB, FFf / GNB);    // (32, 32)
  gemm_wmma<0,1,1,0,1><<<gF, 256, 0, stream>>>(n2_b, w1_b, b1, nullptr, ff1_b, Mrows, FFf, Ee);

  // out = x1 + ff1 @ w2 + b2  (f32)
  gemm_wmma<0,1,0,1,0><<<gE, 256, 0, stream>>>(ff1_b, w2_b, b2, x1, out, Mrows, Ee, FFf);
}
